// MultiHeadAttention_14147622273455
// MI455X (gfx1250) — matrix operand order repaired and checked by reading, not compiled, not run on hardware
//
#include <hip/hip_runtime.h>
#include <hip/hip_bf16.h>

#define S_LEN  4096
#define DM     768
#define HEADS  12
#define DK     64

// log2(e) and score scale 1/sqrt(64) folded together; applied to Q at projection time
#define QSCALE  0.180336880972f          // 0.125 * log2(e)
#define NEG2    (-1.44269504e9f)         // -1e9 * log2(e)  (softmax-equivalent mask value)

typedef __attribute__((ext_vector_type(16))) __bf16 v16bf;
typedef __attribute__((ext_vector_type(8)))  __bf16 v8bf;
typedef __attribute__((ext_vector_type(8)))  float  v8f;

static __device__ __forceinline__ v16bf pack16(v8bf a, v8bf b) {
  v16bf r;
#pragma unroll
  for (int i = 0; i < 8; ++i) { r[i] = a[i]; r[i + 8] = b[i]; }
  return r;
}

// 8 contiguous f32 -> 8 bf16 (two 16B loads, v_cvt_pk_bf16_f32 pairs)
static __device__ __forceinline__ v8bf cvt8(const float* p) {
  const float4 x = ((const float4*)p)[0];
  const float4 y = ((const float4*)p)[1];
  v8bf r;
  r[0] = (__bf16)x.x; r[1] = (__bf16)x.y; r[2] = (__bf16)x.z; r[3] = (__bf16)x.w;
  r[4] = (__bf16)y.x; r[5] = (__bf16)y.y; r[6] = (__bf16)y.z; r[7] = (__bf16)y.w;
  return r;
}

// Y[4096][768] = (X(f32) @ W^T + bias) * outScale, bf16 out.
// One wave owns a 16x64 strip: 1 A fragment feeds 4 WMMAs per k-step.
// STORE_YT stores transposed YT[N][M]; one lane holds 8 consecutive rows -> b128 store.
template <bool STORE_Y, bool STORE_YT>
__global__ void proj_kernel(const float* __restrict__ X, const float* __restrict__ W,
                            const float* __restrict__ bias, float outScale,
                            __bf16* __restrict__ Y, __bf16* __restrict__ YT) {
  const int lane = threadIdx.x & 31;
  const int wave = threadIdx.x >> 5;
  const int lo = lane & 15, hi = lane >> 4;
  const int rowBase = blockIdx.x * 16;
  const int nBase   = blockIdx.y * 256 + wave * 64;

  const float* xrow = X + (size_t)(rowBase + lo) * DM;
  const float* w0   = W + (size_t)(nBase + lo) * DM;

  v8f acc[4] = {};
  for (int kb = 0; kb < DM; kb += 32) {
    // A (16x32): lane=row, K run hi*16..+15
    const v16bf A = pack16(cvt8(xrow + kb + hi * 16), cvt8(xrow + kb + hi * 16 + 8));
#pragma unroll
    for (int t = 0; t < 4; ++t) {
      // B (32x16): lane=col, K run hi*16..+15 ; B[k][n] = W[n][k]
      const float* wr = w0 + (size_t)t * 16 * DM;
      const v16bf B = pack16(cvt8(wr + kb + hi * 16), cvt8(wr + kb + hi * 16 + 8));
      acc[t] = __builtin_amdgcn_wmma_f32_16x16x32_bf16(false, A, false, B,
                                                       (short)0, acc[t], false, false);
    }
  }
#pragma unroll
  for (int t = 0; t < 4; ++t) {
    const int c = nBase + t * 16 + lo;
    const float bv = bias[c];
    if (STORE_Y) {
#pragma unroll
      for (int v = 0; v < 8; ++v)
        Y[(size_t)(rowBase + v + 8 * hi) * DM + c] = (__bf16)((acc[t][v] + bv) * outScale);
    }
    if (STORE_YT) {                       // lane owns rows rowBase+8*hi .. +7: one b128
      v8bf col;
#pragma unroll
      for (int v = 0; v < 8; ++v) col[v] = (__bf16)((acc[t][v] + bv) * outScale);
      *(v8bf*)(YT + (size_t)c * S_LEN + rowBase + 8 * hi) = col;
    }
  }
}

// out[4096][768] = ctx(bf16) @ Wo^T + bo, f32 output; 16x64 strip per wave.
__global__ void outproj_kernel(const __bf16* __restrict__ X, const float* __restrict__ W,
                               const float* __restrict__ bias, float* __restrict__ Y) {
  const int lane = threadIdx.x & 31;
  const int wave = threadIdx.x >> 5;
  const int lo = lane & 15, hi = lane >> 4;
  const int rowBase = blockIdx.x * 16;
  const int nBase   = blockIdx.y * 256 + wave * 64;

  const __bf16* xrow = X + (size_t)(rowBase + lo) * DM;
  const float*  w0   = W + (size_t)(nBase + lo) * DM;

  v8f acc[4] = {};
  for (int kb = 0; kb < DM; kb += 32) {
    const v16bf A = pack16(*(const v8bf*)(xrow + kb + hi * 16),
                           *(const v8bf*)(xrow + kb + hi * 16 + 8));
#pragma unroll
    for (int t = 0; t < 4; ++t) {
      const float* wr = w0 + (size_t)t * 16 * DM;
      const v16bf B = pack16(cvt8(wr + kb + hi * 16), cvt8(wr + kb + hi * 16 + 8));
      acc[t] = __builtin_amdgcn_wmma_f32_16x16x32_bf16(false, A, false, B,
                                                       (short)0, acc[t], false, false);
    }
  }
#pragma unroll
  for (int t = 0; t < 4; ++t) {
    const int c = nBase + t * 16 + lo;
    const float bv = bias[c];
#pragma unroll
    for (int v = 0; v < 8; ++v)
      Y[(size_t)(rowBase + v + 8 * hi) * DM + c] = acc[t][v] + bv;
  }
}

// Flash attention: one wave = 16 query rows of one head; key chunks of 64
// (softmax reductions / O-rescale amortized over 16 WMMAs per chunk).
// Q already carries 0.125*log2e; softmax runs in base 2 (native v_exp_f32).
__global__ void attn_kernel(const __bf16* __restrict__ Qb, const __bf16* __restrict__ Kb,
                            const __bf16* __restrict__ VT, const int* __restrict__ mask,
                            __bf16* __restrict__ ctx) {
  __shared__ __align__(16) __bf16 psc[4 * 16 * 64];   // per-wave P scratch (16x64)
  const int lane = threadIdx.x & 31;
  const int wave = threadIdx.x >> 5;
  const int lo = lane & 15, hi = lane >> 4;
  const int h = blockIdx.y;
  const int qBase = blockIdx.x * 64 + wave * 16;
  __bf16* myp = psc + wave * (16 * 64);

  const __bf16* qrow = Qb + (size_t)(qBase + lo) * DM + h * DK;
  const v16bf qa = pack16(*(const v8bf*)(qrow + hi * 16),
                          *(const v8bf*)(qrow + hi * 16 + 8));
  const v16bf qc = pack16(*(const v8bf*)(qrow + 32 + hi * 16),
                          *(const v8bf*)(qrow + 32 + hi * 16 + 8));

  v8f o[4] = {};
  float m[8], l[8];
#pragma unroll
  for (int v = 0; v < 8; ++v) { m[v] = -1e30f; l[v] = 0.f; }

  for (int kb = 0; kb < S_LEN; kb += 64) {
    // prefetch next chunk's mask row and K row (mask is the dominant HBM stream)
    if (kb + 64 < S_LEN) {
      __builtin_prefetch(mask + (size_t)(qBase + lo) * S_LEN + kb + 64, 0, 0);
      __builtin_prefetch(Kb + (size_t)(kb + 64 + lo) * DM + h * DK, 0, 0);
    }
    // ---- scores: four 16-key tiles, d_k=64 = 2 WMMAs each ----
    v8f s[4];
#pragma unroll
    for (int t = 0; t < 4; ++t) {
      const __bf16* krow = Kb + (size_t)(kb + t * 16 + lo) * DM + h * DK;
      const v16bf kA = pack16(*(const v8bf*)(krow + hi * 16),
                              *(const v8bf*)(krow + hi * 16 + 8));
      const v16bf kB = pack16(*(const v8bf*)(krow + 32 + hi * 16),
                              *(const v8bf*)(krow + 32 + hi * 16 + 8));
      v8f z = {};
      z = __builtin_amdgcn_wmma_f32_16x16x32_bf16(false, qa, false, kA, (short)0, z, false, false);
      z = __builtin_amdgcn_wmma_f32_16x16x32_bf16(false, qc, false, kB, (short)0, z, false, false);
      s[t] = z;
    }
    // ---- mask (scale pre-folded into Q) ----
#pragma unroll
    for (int t = 0; t < 4; ++t) {
      const int col = kb + t * 16 + lo;
#pragma unroll
      for (int v = 0; v < 8; ++v) {
        const int row = qBase + v + 8 * hi;
        if (mask[(size_t)row * S_LEN + col] == 0) s[t][v] = NEG2;
      }
    }
    // ---- online softmax in base 2 (one reduction pass per 64 keys) ----
#pragma unroll
    for (int v = 0; v < 8; ++v) {
      float cm = fmaxf(fmaxf(s[0][v], s[1][v]), fmaxf(s[2][v], s[3][v]));
#pragma unroll
      for (int d = 1; d < 16; d <<= 1) cm = fmaxf(cm, __shfl_xor(cm, d, 32));
      const float mn  = fmaxf(m[v], cm);
      const float fsc = exp2f(m[v] - mn);
      m[v] = mn;
      const float p0 = exp2f(s[0][v] - mn);
      const float p1 = exp2f(s[1][v] - mn);
      const float p2 = exp2f(s[2][v] - mn);
      const float p3 = exp2f(s[3][v] - mn);
      float ps = (p0 + p1) + (p2 + p3);
#pragma unroll
      for (int d = 1; d < 16; d <<= 1) ps += __shfl_xor(ps, d, 32);
      l[v] = l[v] * fsc + ps;
      o[0][v] *= fsc; o[1][v] *= fsc; o[2][v] *= fsc; o[3][v] *= fsc;
      const int rv = v + 8 * hi;                 // C-layout row
      myp[rv * 64 + lo]      = (__bf16)p0;       // C->A transpose via LDS
      myp[rv * 64 + 16 + lo] = (__bf16)p1;
      myp[rv * 64 + 32 + lo] = (__bf16)p2;
      myp[rv * 64 + 48 + lo] = (__bf16)p3;
    }
    // re-read P in A-operand layout (ds ops from one wave are in-order):
    // PA = keys kb..kb+31, PB = keys kb+32..kb+63
    const v16bf PA = pack16(*(const v8bf*)(myp + lo * 64 + hi * 16),
                            *(const v8bf*)(myp + lo * 64 + hi * 16 + 8));
    const v16bf PB = pack16(*(const v8bf*)(myp + lo * 64 + 32 + hi * 16),
                            *(const v8bf*)(myp + lo * 64 + 32 + hi * 16 + 8));
    // ---- ctx += P @ V : 4 output d-tiles x 2 key halves ----
#pragma unroll
    for (int dt = 0; dt < 4; ++dt) {
      const __bf16* vrow = VT + (size_t)(h * DK + dt * 16 + lo) * S_LEN + kb + hi * 16;
      const v16bf Vf0 = pack16(*(const v8bf*)vrow, *(const v8bf*)(vrow + 8));
      const v16bf Vf1 = pack16(*(const v8bf*)(vrow + 32), *(const v8bf*)(vrow + 40));
      o[dt] = __builtin_amdgcn_wmma_f32_16x16x32_bf16(false, PA, false, Vf0,
                                                      (short)0, o[dt], false, false);
      o[dt] = __builtin_amdgcn_wmma_f32_16x16x32_bf16(false, PB, false, Vf1,
                                                      (short)0, o[dt], false, false);
    }
  }
  // ---- normalize, store ctx as bf16 ----
#pragma unroll
  for (int dt = 0; dt < 4; ++dt)
#pragma unroll
    for (int v = 0; v < 8; ++v) {
      const int row = qBase + v + 8 * hi;
      const int col = h * DK + dt * 16 + lo;
      ctx[(size_t)row * DM + col] = (__bf16)(o[dt][v] / l[v]);
    }
}

extern "C" void kernel_launch(void* const* d_in, const int* in_sizes, int n_in,
                              void* d_out, int out_size, void* d_ws, size_t ws_size,
                              hipStream_t stream) {
  const float* q  = (const float*)d_in[0];
  const float* k  = (const float*)d_in[1];
  const float* v  = (const float*)d_in[2];
  const float* Wq = (const float*)d_in[3];
  const float* bq = (const float*)d_in[4];
  const float* Wk = (const float*)d_in[5];
  const float* bk = (const float*)d_in[6];
  const float* Wv = (const float*)d_in[7];
  const float* bv = (const float*)d_in[8];
  const float* Wo = (const float*)d_in[9];
  const float* bo = (const float*)d_in[10];
  const int* mask = (const int*)d_in[11];
  float* out = (float*)d_out;

  const size_t n = (size_t)S_LEN * DM;
  __bf16* Qb  = (__bf16*)d_ws;     // [S][DM] bf16, pre-scaled by 0.125*log2e
  __bf16* Kb  = Qb + n;            // [S][DM] bf16
  __bf16* VT  = Kb + n;            // [DM][S] bf16 (transposed V)
  __bf16* Ctx = VT + n;            // [S][DM] bf16

  const dim3 blk(128);                       // 4 waves
  const dim3 gp(S_LEN / 16, DM / 256);       // 256 x 3 (16x64 strip per wave)
  proj_kernel<true, false><<<gp, blk, 0, stream>>>(q, Wq, bq, QSCALE, Qb, nullptr);
  proj_kernel<true, false><<<gp, blk, 0, stream>>>(k, Wk, bk, 1.0f, Kb, nullptr);
  proj_kernel<false, true><<<gp, blk, 0, stream>>>(v, Wv, bv, 1.0f, nullptr, VT);

  const dim3 ga(S_LEN / 64, HEADS);          // 64 x 12
  attn_kernel<<<ga, blk, 0, stream>>>(Qb, Kb, VT, mask, Ctx);

  outproj_kernel<<<gp, blk, 0, stream>>>(Ctx, Wo, bo, out);
}
